// TailNet_30210799960320
// MI455X (gfx1250) — matrix operand order repaired and checked by reading, not compiled, not run on hardware
//
#include <hip/hip_runtime.h>

typedef __attribute__((ext_vector_type(16))) __bf16 v16bf;
typedef __attribute__((ext_vector_type(8)))  __bf16 v8bf;
typedef __attribute__((ext_vector_type(8)))  float  v8f;

#define Bx   16
#define Tt   64
#define Hh   256
#define Ww   256
#define Cc   3
#define PADp 3
#define TCH  16               // t per chunk (K sub-tile = 2 taps x 16 t = 32)
#define NCH  (Tt / TCH)       // 4 chunks
#define TILE_W 128            // w per workgroup (8 waves x 16)
#define SLAB_W 134            // 128 + 6 halo
#define SLAB_WP 136           // padded row length
#define NP   50               // 49 taps padded to 50 (p=49 row is zeros)

__global__ __launch_bounds__(256)
void conv_dict_wmma(const float* __restrict__ x,
                    const float* __restrict__ d,
                    float* __restrict__ out) {
  // xs[kh][w][t]: a lane's B fragment = 32 contiguous bytes (t fastest)
  // dsd[c][p][t]: c padded to 4, row c=3 all zeros (M-padding), p=49 all zeros
  __shared__ __align__(32) __bf16 xs[7][SLAB_WP][TCH];
  __shared__ __align__(32) __bf16 dsd[4][NP][TCH];

  const int gid   = blockIdx.x;            // b*512 + h*2 + wchunk
  const int b     = gid >> 9;
  const int rem   = gid & 511;
  const int h     = rem >> 1;
  const int wbase = (rem & 1) * TILE_W;

  const int tid  = threadIdx.x;
  const int lane = tid & 31;
  const int wave = tid >> 5;
  const int n    = lane & 15;              // N (w offset) / M (c row)
  const int hi   = lane >> 4;              // K half select

  // A rows >= 3 read the zero row c=3 -> unconditional loads, exact zero pad
  const int m_cl = (n < Cc) ? n : Cc;
  const __bf16* aBase = &dsd[m_cl][0][0] + hi * TCH;     // + p*TCH (immediates)
  const __bf16* bBase = &xs[0][wave * 16 + n][0];        // + (kh*136+kw)*TCH

  v8f acc = {0.f, 0.f, 0.f, 0.f, 0.f, 0.f, 0.f, 0.f};

  for (int ch = 0; ch < NCH; ++ch) {
    const int t0 = ch * TCH;

    // ---- stage x slab: fp32 -> bf16, [kh][w][t] ----
    for (int i = tid; i < TCH * 7 * SLAB_W; i += 256) {
      int t  = i / (7 * SLAB_W);
      int r  = i % (7 * SLAB_W);
      int kh = r / SLAB_W;
      int l  = r % SLAB_W;
      int row = (h + kh - PADp) & (Hh - 1);              // circular wrap
      int col = (wbase + l - PADp) & (Ww - 1);
      xs[kh][l][t] =
          (__bf16)x[(((size_t)b * Tt + (t0 + t)) * Hh + row) * Ww + col];
    }
    // ---- stage d chunk: [c][p][t], c=3 and p=49 zero-filled ----
    for (int j = tid; j < 4 * NP * TCH; j += 256) {
      int c = j / (NP * TCH);
      int r = j % (NP * TCH);
      int p = r / TCH;
      int t = r % TCH;
      __bf16 v = (__bf16)0.f;
      if (c < Cc && p < 49)
        v = (__bf16)d[(((size_t)b * Cc + c) * Tt + (t0 + t)) * 49 + p];
      dsd[c][p][t] = v;
    }
    if (ch + 1 < NCH)
      __builtin_prefetch(&x[(((size_t)b * Tt + (t0 + TCH)) * Hh) * Ww], 0, 1);

    __syncthreads();

    // ---- 25 WMMAs, all LDS offsets immediate after unroll ----
#pragma unroll
    for (int pb = 0; pb < NP; pb += 2) {
      const int p1    = (pb + 1 > 48) ? 48 : pb + 1;     // addr clamp; A is 0
      const int offB0 = ((pb / 7) * SLAB_WP + (pb % 7)) * TCH;   // constants
      const int offB1 = ((p1 / 7) * SLAB_WP + (p1 % 7)) * TCH;
      const int offB  = hi ? offB1 : offB0;              // one v_cndmask

      v8bf alo = *(const v8bf*)(aBase + pb * TCH);           // ds_load_b128
      v8bf ahi = *(const v8bf*)(aBase + pb * TCH + 8);       // ds_load_b128
      v16bf a  = __builtin_shufflevector(alo, ahi,
                   0,1,2,3,4,5,6,7,8,9,10,11,12,13,14,15);
      v16bf bb = *(const v16bf*)(bBase + offB);              // 2x ds_load_b128

      acc = __builtin_amdgcn_wmma_f32_16x16x32_bf16(
          false, a, false, bb, (short)0, acc, false, false);
    }
    __syncthreads();
  }

  // C/D: lanes 0-15 hold N=0..15, VGPR r = row M=r. Rows 0..2 are valid output.
  if (lane < 16) {
    const int w = wbase + wave * 16 + n;
#pragma unroll
    for (int r = 0; r < Cc; ++r)
      out[(((size_t)b * Cc + r) * Hh + h) * Ww + w] = acc[r];
  }
}

extern "C" void kernel_launch(void* const* d_in, const int* in_sizes, int n_in,
                              void* d_out, int out_size, void* d_ws, size_t ws_size,
                              hipStream_t stream) {
  const float* x = (const float*)d_in[0];   // [16,64,256,256] fp32
  const float* d = (const float*)d_in[1];   // [16,3,64,7,7]   fp32
  float* out = (float*)d_out;               // [16,3,256,256]  fp32
  (void)in_sizes; (void)n_in; (void)d_ws; (void)ws_size;

  dim3 grid(Bx * Hh * (Ww / TILE_W));       // 8192 workgroups
  dim3 block(256);                          // 8 waves (wave32)
  conv_dict_wmma<<<grid, block, 0, stream>>>(x, d, out);
}
